// E44DiagonalW_47656957116868
// MI455X (gfx1250) — matrix operand order repaired and checked by reading, not compiled, not run on hardware
//
#include <hip/hip_runtime.h>
#include <hip/hip_bf16.h>

// MI455X (gfx1250): x -> f16 once; two f16-WMMA GEMMs with async global->LDS
// double-buffered staging (ASYNCcnt path); chunk-parallel affine scan in f32.
// Roofline: ~0.6 GB HBM traffic -> ~30us floor at 23.3 TB/s; f16 WMMA keeps
// the 2x137 GFLOP GEMMs at that floor (f32 16x16x4 path would be 8x slower).

typedef _Float16 v4h  __attribute__((ext_vector_type(4)));
typedef _Float16 v8h  __attribute__((ext_vector_type(8)));
typedef _Float16 v16h __attribute__((ext_vector_type(16)));
typedef float    v8f  __attribute__((ext_vector_type(8)));

#define BM 128
#define BN 128
#define TK 32
#define LDSS 48   // LDS row stride in f16 (32 data + 16 pad) -> 96B rows, 16B-aligned frags

// ---------------- f32 -> f16 bulk conversion ----------------
__global__ void f32_to_f16_kernel(const float* __restrict__ src,
                                  _Float16* __restrict__ dst, int n4)
{
    int i = blockIdx.x * blockDim.x + threadIdx.x;
    if (i >= n4) return;
    float4 v = ((const float4*)src)[i];
    v4h h;
    h[0] = (_Float16)v.x; h[1] = (_Float16)v.y;
    h[2] = (_Float16)v.z; h[3] = (_Float16)v.w;
    ((v4h*)dst)[i] = h;
}

// C = act( A[M,K] @ W[N,K]^T ), A/W f16. mode=1: silu -> f16 Ch; mode=0: f32 Cf.
__global__ __launch_bounds__(256) void gemm_f16wmma_kernel(
    const _Float16* __restrict__ A, const _Float16* __restrict__ W,
    float* __restrict__ Cf, _Float16* __restrict__ Ch,
    int M, int N, int K, int mode)
{
    __shared__ _Float16 Ash[2][BM * LDSS];
    __shared__ _Float16 Bsh[2][BN * LDSS];

    const int tid   = threadIdx.x;
    const int lane  = tid & 31;
    const int wid   = tid >> 5;
    const int waveM = wid >> 2;          // 0..1 -> 64 rows each
    const int waveN = wid & 3;           // 0..3 -> 32 cols each
    const int bm0   = blockIdx.x * BM;
    const int bn0   = blockIdx.y * BN;

    const _Float16* Ablk = A + (size_t)bm0 * K;
    const _Float16* Wblk = W + (size_t)bn0 * K;

    // Per K-stage each thread moves 2x16B of A and 2x16B of B (128x32 f16 tiles).
    // chunk c = rep*256+tid: row = c>>2, col8 = (c&3)*8 (f16 units).
    const int r0 = (0 * 256 + tid) >> 2, c0 = ((0 * 256 + tid) & 3) << 3;
    const int r1 = (1 * 256 + tid) >> 2, c1 = ((1 * 256 + tid) & 3) << 3;

    v8f acc[4][2];
#pragma unroll
    for (int i = 0; i < 4; ++i)
#pragma unroll
        for (int j = 0; j < 2; ++j)
#pragma unroll
            for (int r = 0; r < 8; ++r) acc[i][j][r] = 0.0f;

    const int nStages = K / TK;

    // async-issue one full stage (4x global_load_async_to_lds_b128) into buffer `buf`
    auto issue_stage = [&](int buf, int k0) {
        unsigned l0 = (unsigned)(uintptr_t)&Ash[buf][r0 * LDSS + c0];
        unsigned l1 = (unsigned)(uintptr_t)&Ash[buf][r1 * LDSS + c1];
        unsigned l2 = (unsigned)(uintptr_t)&Bsh[buf][r0 * LDSS + c0];
        unsigned l3 = (unsigned)(uintptr_t)&Bsh[buf][r1 * LDSS + c1];
        unsigned long long g0 = (unsigned long long)(uintptr_t)(Ablk + (size_t)r0 * K + k0 + c0);
        unsigned long long g1 = (unsigned long long)(uintptr_t)(Ablk + (size_t)r1 * K + k0 + c1);
        unsigned long long g2 = (unsigned long long)(uintptr_t)(Wblk + (size_t)r0 * K + k0 + c0);
        unsigned long long g3 = (unsigned long long)(uintptr_t)(Wblk + (size_t)r1 * K + k0 + c1);
        asm volatile("global_load_async_to_lds_b128 %0, %1, off" :: "v"(l0), "v"(g0) : "memory");
        asm volatile("global_load_async_to_lds_b128 %0, %1, off" :: "v"(l1), "v"(g1) : "memory");
        asm volatile("global_load_async_to_lds_b128 %0, %1, off" :: "v"(l2), "v"(g2) : "memory");
        asm volatile("global_load_async_to_lds_b128 %0, %1, off" :: "v"(l3), "v"(g3) : "memory");
    };

    issue_stage(0, 0);                            // prologue: stage 0 in flight

    for (int s = 0; s < nStages; ++s) {
        const int buf = s & 1;
        if (s + 1 < nStages) {
            issue_stage(buf ^ 1, (s + 1) * TK);   // overlap next stage's DMA
            asm volatile("s_wait_asynccnt 0x4" ::: "memory");  // stage s landed
        } else {
            asm volatile("s_wait_asynccnt 0x0" ::: "memory");
        }
        __syncthreads();

        // ---- fragments straight in ISA layout, 2x ds_load_b128 each ----
        // A 16x32 f16: lanes<16: M=lane, K 0..7 (V0-3) & 16..23 (V4-7);
        //              lanes>=16: K 8..15 & 24..31.
        const int mrb = waveM * 64 + (lane & 15);
        const int kLo = (lane >> 4) << 3;         // 0 or 8
        v16h af[4];
#pragma unroll
        for (int i = 0; i < 4; ++i) {
            const _Float16* p = &Ash[buf][(mrb + i * 16) * LDSS];
            v8h lo = *(const v8h*)(p + kLo);
            v8h hi = *(const v8h*)(p + 16 + kLo);
            af[i] = __builtin_shufflevector(lo, hi, 0,1,2,3,4,5,6,7,8,9,10,11,12,13,14,15);
        }
        // B 32x16 f16: lanes<16: N=lane, K 0..7 & 16..23; lanes>=16: K 8..15 & 24..31.
        const int nrb = waveN * 32 + (lane & 15);
        const int kB  = (lane >> 4) << 3;         // 0 or 8
        v16h bf[2];
#pragma unroll
        for (int j = 0; j < 2; ++j) {
            const _Float16* p = &Bsh[buf][(nrb + j * 16) * LDSS + kB];
            v8h lo = *(const v8h*)(p);
            v8h hi = *(const v8h*)(p + 16);
            bf[j] = __builtin_shufflevector(lo, hi, 0,1,2,3,4,5,6,7,8,9,10,11,12,13,14,15);
        }
#pragma unroll
        for (int i = 0; i < 4; ++i)
#pragma unroll
            for (int j = 0; j < 2; ++j)
                acc[i][j] = __builtin_amdgcn_wmma_f32_16x16x32_f16(
                    false, af[i], false, bf[j], (short)0, acc[i][j], false, false);
        __syncthreads();   // protect buf before stage s+2 overwrites it
    }

    // ---- writeback: C/D layout: VGPR r -> M = r + 8*(lane>=16), N = lane&15 ----
#pragma unroll
    for (int i = 0; i < 4; ++i)
#pragma unroll
        for (int j = 0; j < 2; ++j) {
            int m0 = bm0 + waveM * 64 + i * 16 + ((lane >> 4) << 3);
            int n  = bn0 + waveN * 32 + j * 16 + (lane & 15);
#pragma unroll
            for (int r = 0; r < 8; ++r) {
                float v = acc[i][j][r];
                if (mode) {
                    v = v / (1.0f + __expf(-v));            // silu
                    Ch[(size_t)(m0 + r) * N + n] = (_Float16)v;
                } else {
                    Cf[(size_t)(m0 + r) * N + n] = v;
                }
            }
        }
}

// ---- pass 1: per-chunk recurrence with h_in = 0, record chunk endpoint ----
__global__ void scan_pass1_kernel(const _Float16* __restrict__ xp,
                                  const float* __restrict__ log_d,
                                  const float* __restrict__ bias,
                                  float* __restrict__ chunkC,
                                  int Bsz, int T, int D, int CH, int L)
{
    int tid = blockIdx.x * blockDim.x + threadIdx.x;
    if (tid >= Bsz * CH * D) return;
    int e = tid % D, rem = tid / D;
    int c = rem % CH, b = rem / CH;
    float d  = 1.0f / (1.0f + __expf(-log_d[e]));
    float bv = bias[e];
    float h  = 0.0f;
    const _Float16* px = xp + ((size_t)(b * T + c * L)) * D + e;
    for (int i = 0; i < L; ++i) { h = d * ((float)*px + h) + bv; px += D; }
    chunkC[((size_t)b * CH + c) * D + e] = h;
}

// ---- carry: h_in(c+1) = d^L * h_in(c) + chunkC(c); also emits h_final ----
__global__ void scan_carry_kernel(const float* __restrict__ h0,
                                  const float* __restrict__ log_d,
                                  const float* __restrict__ chunkC,
                                  float* __restrict__ chunkH,
                                  float* __restrict__ h_final,
                                  int Bsz, int D, int CH, int L)
{
    int tid = blockIdx.x * blockDim.x + threadIdx.x;
    if (tid >= Bsz * D) return;
    int e = tid % D, b = tid / D;
    float d  = 1.0f / (1.0f + __expf(-log_d[e]));
    float dL = __powf(d, (float)L);
    float h  = h0[(size_t)b * D + e];
    for (int c = 0; c < CH; ++c) {
        size_t idx = ((size_t)b * CH + c) * D + e;
        chunkH[idx] = h;
        h = dL * h + chunkC[idx];
    }
    h_final[(size_t)b * D + e] = h;
}

// ---- pass 2: redo chunk with correct h_in, write gated f16 output in place ----
__global__ void scan_pass2_kernel(_Float16* __restrict__ xp,
                                  const float* __restrict__ log_d,
                                  const float* __restrict__ bias,
                                  const float* __restrict__ chunkH,
                                  int Bsz, int T, int D, int CH, int L)
{
    int tid = blockIdx.x * blockDim.x + threadIdx.x;
    if (tid >= Bsz * CH * D) return;
    int e = tid % D, rem = tid / D;
    int c = rem % CH, b = rem / CH;
    float d  = 1.0f / (1.0f + __expf(-log_d[e]));
    float bv = bias[e];
    float h  = chunkH[((size_t)b * CH + c) * D + e];
    _Float16* px = xp + ((size_t)(b * T + c * L)) * D + e;
    for (int i = 0; i < L; ++i) {
        h = d * ((float)*px + h) + bv;
        float s = 1.0f / (1.0f + __expf(-h));
        *px = (_Float16)(h * h * s);              // h * silu(h)
        px += D;
    }
}

extern "C" void kernel_launch(void* const* d_in, const int* in_sizes, int n_in,
                              void* d_out, int out_size, void* d_ws, size_t ws_size,
                              hipStream_t stream) {
    const float* x     = (const float*)d_in[0];
    const float* h0    = (const float*)d_in[1];
    const float* W_in  = (const float*)d_in[2];
    const float* W_out = (const float*)d_in[3];
    const float* log_d = (const float*)d_in[4];
    const float* bvec  = (const float*)d_in[5];
    float* out = (float*)d_out;

    const int B = 4, T = 4096, DIM = 2048, D = 2048;
    const int CH = 32, L = 128;                   // CH*L == T
    const int M = B * T;                          // 16384

    // workspace layout (f16 buffers first, all 16B aligned)
    _Float16* xph = (_Float16*)d_ws;                        // M*D   (64 MiB) xp / cell
    _Float16* xh  = xph + (size_t)M * D;                    // M*DIM (64 MiB) f16(x)
    _Float16* Wh1 = xh  + (size_t)M * DIM;                  // D*DIM (8 MiB)
    _Float16* Wh2 = Wh1 + (size_t)D * DIM;                  // DIM*D (8 MiB)
    float* chunkC = (float*)(Wh2 + (size_t)DIM * D);        // B*CH*D
    float* chunkH = chunkC + (size_t)B * CH * D;            // B*CH*D
    float* h_fin  = out + (size_t)M * DIM;                  // tail of d_out

    dim3 blk(256);

    int n4x = (M * DIM) / 4, n4w = (D * DIM) / 4;
    f32_to_f16_kernel<<<(n4x + 255) / 256, blk, 0, stream>>>(x, xh, n4x);
    f32_to_f16_kernel<<<(n4w + 255) / 256, blk, 0, stream>>>(W_in, Wh1, n4w);
    f32_to_f16_kernel<<<(n4w + 255) / 256, blk, 0, stream>>>(W_out, Wh2, n4w);

    dim3 g1(M / BM, D / BN);
    gemm_f16wmma_kernel<<<g1, blk, 0, stream>>>(xh, Wh1, nullptr, xph, M, D, DIM, 1);

    int tot1 = B * CH * D;
    scan_pass1_kernel<<<(tot1 + 255) / 256, blk, 0, stream>>>(xph, log_d, bvec, chunkC,
                                                              B, T, D, CH, L);
    int tot2 = B * D;
    scan_carry_kernel<<<(tot2 + 255) / 256, blk, 0, stream>>>(h0, log_d, chunkC, chunkH,
                                                              h_fin, B, D, CH, L);
    scan_pass2_kernel<<<(tot1 + 255) / 256, blk, 0, stream>>>(xph, log_d, bvec, chunkH,
                                                              B, T, D, CH, L);

    dim3 g2(M / BM, DIM / BN);
    gemm_f16wmma_kernel<<<g2, blk, 0, stream>>>(xph, Wh2, out, nullptr, M, DIM, D, 0);
}
